// PatchTransformer_43490838840065
// MI455X (gfx1250) — matrix operand order repaired and checked by reading, not compiled, not run on hardware
//
#include <hip/hip_runtime.h>

typedef __attribute__((ext_vector_type(16))) _Float16 v16h;
typedef __attribute__((ext_vector_type(8)))  float    v8f;
typedef int v4i_ __attribute__((vector_size(16)));   // matches builtin param type

#define BATCH 16
#define TGT   64
#define PSZ   128
#define IMH   512
#define IMW   512
#define PLANE (IMH * IMW)

#if defined(__has_builtin)
#  if __has_builtin(__builtin_amdgcn_global_load_async_to_lds_b128)
#    define HAVE_ASYNC_LDS 1
#  endif
#endif
#ifndef HAVE_ASYNC_LDS
#  define HAVE_ASYNC_LDS 0
#endif

// One wave per 16-pixel tile. 8 waves per 256-thread block.
// vals/covs for all 64 targets are reduced to 16 batches via
// one-hot x vals matmul on V_WMMA_F32_16X16X32_F16.
__global__ __launch_bounds__(256) void patch_composite_wmma(
    const float* __restrict__ patch,
    const float* __restrict__ targets,
    const float* __restrict__ angles,
    const float* __restrict__ temps,
    float* __restrict__ out)
{
    __shared__ float sPatch[PSZ * PSZ];   // 64 KB raw patch (temp-add applied per tap)
    __shared__ float sParm[TGT][8];       // cx, cy, cos, sin, scale, add, idx

    const int tid = threadIdx.x;

#if HAVE_ASYNC_LDS
    { // async DMA patch -> LDS: 16 x b128 per thread, no VGPR round trip
        const unsigned long long gbase = (unsigned long long)patch;
        const unsigned int lbase = (unsigned int)(unsigned long long)
            (__attribute__((address_space(3))) char*)
            (__attribute__((address_space(3))) void*)sPatch;
        #pragma unroll
        for (int i = 0; i < (PSZ * PSZ * 4 / 16) / 256; ++i) {
            const unsigned int off = (unsigned int)((tid + i * 256) * 16);
            __builtin_amdgcn_global_load_async_to_lds_b128(
                (__attribute__((address_space(1))) v4i_*)(gbase + off),
                (__attribute__((address_space(3))) v4i_*)(lbase + off),
                0, 0);
        }
    }
#else
    { // fallback: stage patch via float4 copies
        const float4* __restrict__ g4 = (const float4*)patch;
        float4* s4 = (float4*)sPatch;
        #pragma unroll
        for (int i = 0; i < (PSZ * PSZ / 4) / 256; ++i)
            s4[tid + i * 256] = g4[tid + i * 256];
    }
#endif

    if (tid < TGT) { // per-target affine params
        const float* tg = targets + tid * 6;
        float ps  = tg[5] * (float)IMH * 0.2f;           // RATIO_H
        float th  = angles[tid] * 0.017453292519943295f; // deg->rad
        float c   = cosf(th);
        float s   = sinf(th);
        float ext = ps * (fabsf(c) + fabsf(s));          // rotated bbox side
        sParm[tid][0] = tg[2] * (float)IMW;              // cx
        sParm[tid][1] = tg[3] * (float)IMH - 0.2f * ext; // cy
        sParm[tid][2] = c;
        sParm[tid][3] = s;
        sParm[tid][4] = (float)(PSZ - 1) / fmaxf(ps - 1.0f, 1.0f); // align_corners inv scale
        sParm[tid][5] = 0.1f * temps[tid];               // TEMP_FACTOR * seed
        sParm[tid][6] = tg[0];                           // img_idx
    }

#if HAVE_ASYNC_LDS
#  if __has_builtin(__builtin_amdgcn_s_wait_asynccnt)
    __builtin_amdgcn_s_wait_asynccnt(0);
#  else
    asm volatile("s_wait_asynccnt 0" ::: "memory");
#  endif
#endif
    __syncthreads();

    const int lane = tid & 31;          // wave32
    const int wv   = tid >> 5;
    const int n    = lane & 15;         // pixel column / batch row (M==N==lane%16)
    const int hsel = lane >> 4;         // lane half selector

    const int tile = blockIdx.x * 8 + wv;   // 16384 tiles of 16 px
    const int y    = tile >> 5;             // 32 tiles per image row
    const int x0   = (tile & 31) << 4;
    const float fx = (float)(x0 + n);
    const float fy = (float)y;

    v8f cVal = {};
    v8f cCov = {};

    #pragma unroll
    for (int chunk = 0; chunk < 2; ++chunk) {
        const int tbase = chunk * 32;
        v16h aOne, bVal, bCov;
        #pragma unroll
        for (int e = 0; e < 16; ++e) {
            // A-matrix 16x32: element e of lane-half h -> K = e + 16*h (matches B)
            const int ka = e + 16 * hsel;
            // B-matrix 32x16, contiguous K order used here (same as A): lanes 0-15 hold K=0..15, lanes 16-31 K=16..31
            const int kb = e + 16 * hsel;

            // A: one-hot (img_idx == batch row)
            const int ta = tbase + ka;
            aOne[e] = ((int)sParm[ta][6] == n) ? (_Float16)1.0f : (_Float16)0.0f;

            // B: bilinear sample of target tb at this pixel
            const int tb = tbase + kb;
            const float cx = sParm[tb][0], cy = sParm[tb][1];
            const float c  = sParm[tb][2], s  = sParm[tb][3];
            const float sc = sParm[tb][4], ad = sParm[tb][5];
            const float dx = fx - cx, dy = fy - cy;
            const float px = ( c * dx + s * dy) * sc + 63.5f; // half = (P-1)/2
            const float py = (-s * dx + c * dy) * sc + 63.5f;
            const float xf = floorf(px), yf = floorf(py);
            const int ix0 = (int)xf, iy0 = (int)yf;
            const float wx = px - xf, wy = py - yf;
            float val = 0.f, cov = 0.f;
            #pragma unroll
            for (int t4 = 0; t4 < 4; ++t4) {
                const int diy = t4 >> 1, dix = t4 & 1;
                const int iy = iy0 + diy, ix = ix0 + dix;
                const bool ok = (iy >= 0) & (iy < PSZ) & (ix >= 0) & (ix < PSZ);
                const float w = (diy ? wy : 1.f - wy) * (dix ? wx : 1.f - wx);
                const int cyc = iy < 0 ? 0 : (iy > PSZ - 1 ? PSZ - 1 : iy);
                const int cxc = ix < 0 ? 0 : (ix > PSZ - 1 ? PSZ - 1 : ix);
                float tv = sPatch[cyc * PSZ + cxc] + ad;       // temp add
                tv = fminf(fmaxf(tv, 0.f), 1.f);               // clamp BEFORE interp
                val += ok ? tv * w : 0.f;
                cov += ok ? w : 0.f;
            }
            bVal[e] = (_Float16)val;
            bCov[e] = (_Float16)cov;
        }
        // D[16 batches x 16 pixels] += onehot[16x32] x vals[32x16]
        cVal = __builtin_amdgcn_wmma_f32_16x16x32_f16(
            false, aOne, false, bVal, (short)0, cVal, false, false);
        cCov = __builtin_amdgcn_wmma_f32_16x16x32_f16(
            false, aOne, false, bCov, (short)0, cCov, false, false);
    }

    // C/D layout: VGPR r, lane half h -> batch M = r + 8*h; pixel N = lane%16.
    // Broadcast over 3 channels; mask_tf follows patch_tf in d_out.
    const int maskOff = BATCH * 3 * PLANE;
    const int pixOff  = y * IMW + x0 + n;
    #pragma unroll
    for (int r = 0; r < 8; ++r) {
        const int mb   = r + 8 * hsel;
        const int base = mb * 3 * PLANE + pixOff;
        const float pv = cVal[r];
        const float mv = cCov[r];
        #pragma unroll
        for (int ch = 0; ch < 3; ++ch) {
            out[base + ch * PLANE]           = pv;
            out[maskOff + base + ch * PLANE] = mv;
        }
    }
}

extern "C" void kernel_launch(void* const* d_in, const int* in_sizes, int n_in,
                              void* d_out, int out_size, void* d_ws, size_t ws_size,
                              hipStream_t stream) {
    const float* patch   = (const float*)d_in[0]; // [1,128,128]
    const float* targets = (const float*)d_in[1]; // [64,6]
    // d_in[2] = imgs: only its shape matters, values unused by the reference output
    const float* angles  = (const float*)d_in[3]; // [64]
    const float* temps   = (const float*)d_in[4]; // [64]
    float* out = (float*)d_out;                   // patch_tf ++ mask_tf, f32

    const int tiles = (IMH * IMW) / 16;  // 16384
    dim3 grid(tiles / 8);                // 8 waves (tiles) per block
    dim3 block(256);
    hipLaunchKernelGGL(patch_composite_wmma, grid, block, 0, stream,
                       patch, targets, angles, temps, out);
}
